// Tenet_52183852646439
// MI455X (gfx1250) — matrix operand order repaired and checked by reading, not compiled, not run on hardware
//
#include <hip/hip_runtime.h>
#include <hip/hip_bf16.h>
#include <math.h>

// ---------------------------------------------------------------------------
// GNN forward for MI455X (gfx1250, wave32, WMMA).
// - all big matmuls: v_wmma_f32_16x16x32_bf16 (f32 accum, bf16 operands)
// - flash attention hot loop feeds LDS with GLOBAL_LOAD_ASYNC_TO_LDS_B128
//   from pre-converted bf16 tensors (ASYNCcnt-tracked, no VGPR round trip)
// - bf16 conversion via native casts packed 2/dword (v_cvt_pk_bf16_f32)
// Workspace: ~84 MB.
// ---------------------------------------------------------------------------

#define NN 8192
#define EE 262144
#define FIN 128
#define HD 256
#define NC 10
#define NG 64
#define BN_S 0.9999950000374997f /* 1/sqrt(1+1e-5) */

typedef __attribute__((ext_vector_type(16))) __bf16 bf16x16;
typedef __attribute__((ext_vector_type(8)))  float  f32x8;

union FragU { uint4 q[2]; bf16x16 v; };

__device__ __forceinline__ unsigned short bf1(float x) {
    union { __bf16 h; unsigned short u; } z;
    z.h = (__bf16)x;
    return z.u;
}

__device__ __forceinline__ unsigned int pk_bf16(float a, float b) {
    union { __bf16 h[2]; unsigned int u; } z;
    z.h[0] = (__bf16)a;
    z.h[1] = (__bf16)b;
    return z.u;
}

__device__ __forceinline__ f32x8 wmma_bf16(bf16x16 a, bf16x16 b, f32x8 c) {
    return __builtin_amdgcn_wmma_f32_16x16x32_bf16(false, a, false, b,
                                                   (short)0, c, false, false);
}

// A fragment (16xK, 16-bit): lane<16 -> M=lane, K=k0+{0..7,16..23};
// lane>=16 -> M=lane-16, K=k0+{8..15,24..31}. LDS [M][K], ld in ushorts.
__device__ __forceinline__ bf16x16 frag_a(const unsigned short* lds, int rowBase,
                                          int ld, int k0, int lane) {
    int m = lane & 15, kh = lane >> 4;
    const uint4* p = (const uint4*)(lds + (size_t)(rowBase + m) * ld);
    int b = k0 >> 3;
    FragU f;
    f.q[0] = p[b + kh];
    f.q[1] = p[b + 2 + kh];
    return f.v;
}

// B fragment (Kx16, 16-bit) from TRANSPOSED LDS storage [N][K]:
// lane<16: N=lane, K=k0+{0..7,16..23}; lane>=16: N=lane-16, K=k0+{8..15,24..31}.
__device__ __forceinline__ bf16x16 frag_b(const unsigned short* lds, int nBase,
                                          int ld, int k0, int lane) {
    int n = lane & 15, kh = lane >> 4;
    const uint4* p = (const uint4*)(lds + (size_t)(nBase + n) * ld);
    int b = (k0 >> 3) + kh;
    FragU f;
    f.q[0] = p[b];
    f.q[1] = p[b + 2];
    return f.v;
}

// CDNA5 async copy: 16 bytes/lane, global -> LDS, tracked by ASYNCcnt.
__device__ __forceinline__ void async_load_b128(unsigned lds_off, const void* gaddr) {
    asm volatile("global_load_async_to_lds_b128 %0, %1, off"
                 :: "v"(lds_off), "v"((unsigned long long)(size_t)gaddr)
                 : "memory");
}
__device__ __forceinline__ void wait_async0() {
    asm volatile("s_wait_asynccnt 0x0" ::: "memory");
}

// ---------------------------------------------------------------------------
// small utility kernels
// ---------------------------------------------------------------------------
__global__ void k_fill(float* __restrict__ p, float v, int n) {
    int i = blockIdx.x * blockDim.x + threadIdx.x;
    if (i < n) p[i] = v;
}

__global__ void k_deg_edges(const int* __restrict__ dst, float* __restrict__ deg, int e) {
    int i = blockIdx.x * blockDim.x + threadIdx.x;
    if (i < e) atomicAdd(&deg[dst[i]], 1.0f);
}

__global__ void k_rsqrt(float* __restrict__ d, int n) {
    int i = blockIdx.x * blockDim.x + threadIdx.x;
    if (i < n) d[i] = rsqrtf(fmaxf(d[i], 1.0f));
}

__global__ void k_selfinit(const float* __restrict__ X, const float* __restrict__ dinv,
                           float* __restrict__ agg, int n, int F) {
    int idx = blockIdx.x * blockDim.x + threadIdx.x;
    if (idx >= n * F) return;
    int i = idx / F;
    float w = dinv[i];
    agg[idx] = X[idx] * w * w;
}

__global__ void k_scatter(const int* __restrict__ src, const int* __restrict__ dst,
                          const float* __restrict__ dinv, const float* __restrict__ X,
                          float* __restrict__ agg, int e, int F4) {
    int idx = blockIdx.x * blockDim.x + threadIdx.x;
    if (idx >= e * F4) return;
    int ed = idx / F4, c = (idx - ed * F4) * 4;
    int s = src[ed], d = dst[ed];
    float w = dinv[s] * dinv[d];
    int F = F4 * 4;
    float4 xv = *(const float4*)(X + (size_t)s * F + c);
    float* ap = agg + (size_t)d * F + c;
    atomicAdd(ap + 0, xv.x * w);
    atomicAdd(ap + 1, xv.y * w);
    atomicAdd(ap + 2, xv.z * w);
    atomicAdd(ap + 3, xv.w * w);
}

// f32 -> bf16 row-major (paired) and transposed converters
__global__ void k_cvt_pk(const float* __restrict__ in, unsigned short* __restrict__ out,
                         int npairs) {
    int i = blockIdx.x * blockDim.x + threadIdx.x;
    if (i >= npairs) return;
    float2 v = ((const float2*)in)[i];
    ((unsigned int*)out)[i] = pk_bf16(v.x, v.y);
}
__global__ void k_cvt_t(const float* __restrict__ in, unsigned short* __restrict__ out) {
    int idx = blockIdx.x * blockDim.x + threadIdx.x; // NN*HD
    int r = idx >> 8, h = idx & 255;
    out[(size_t)h * NN + r] = bf1(in[idx]);
}

// ---------------------------------------------------------------------------
// WMMA GEMM: out[M,N] = epi(A[M,K] @ W[K,N] + bias); tiles 128(M)x64(N), BK=32.
// flags bit0 = BatchNorm (g,be), bit1 = ReLU.
// 8 waves: wave w -> rows 16w..16w+15, all four 16-col subtiles.
// ---------------------------------------------------------------------------
__global__ __launch_bounds__(256) void k_gemm(const float* __restrict__ A,
                                              const float* __restrict__ W,
                                              const float* __restrict__ bias,
                                              const float* __restrict__ gam,
                                              const float* __restrict__ bet,
                                              float* __restrict__ out,
                                              int M, int K, int N, int flags) {
    const int LDA = 40, LDB = 40;
    __shared__ unsigned short Al[128 * 40];
    __shared__ unsigned short Bl[64 * 40];
    int t = threadIdx.x, lane = t & 31, w = t >> 5;
    int m0 = blockIdx.x * 128, n0 = blockIdx.y * 64;

    f32x8 acc[4];
#pragma unroll
    for (int s = 0; s < 4; ++s) acc[s] = (f32x8){0.f,0.f,0.f,0.f,0.f,0.f,0.f,0.f};

    for (int k0 = 0; k0 < K; k0 += 32) {
#pragma unroll
        for (int i = 0; i < 8; ++i) { // A tile 128x32, 2 elems/thread/iter
            int pidx = i * 256 + t;
            int r = pidx >> 4, c = (pidx & 15) * 2;
            float2 v = *(const float2*)(A + (size_t)(m0 + r) * K + k0 + c);
            *(unsigned int*)&Al[r * LDA + c] = pk_bf16(v.x, v.y);
        }
#pragma unroll
        for (int i = 0; i < 8; ++i) { // B tile 32x64 -> transposed [n][k]
            int idx = i * 256 + t;
            int kk = idx >> 6, n = idx & 63;
            Bl[n * LDB + kk] = bf1(W[(size_t)(k0 + kk) * N + n0 + n]);
        }
        if (k0 + 32 < K) // prefetch next A k-tile (global_prefetch_b8)
            __builtin_prefetch(A + (size_t)(m0 + (t >> 1)) * K + k0 + 32, 0, 1);
        __syncthreads();
        bf16x16 fa = frag_a(Al, 16 * w, LDA, 0, lane);
#pragma unroll
        for (int s = 0; s < 4; ++s) {
            bf16x16 fb = frag_b(Bl, 16 * s, LDB, 0, lane);
            acc[s] = wmma_bf16(fa, fb, acc[s]);
        }
        __syncthreads();
    }

    int doBN = flags & 1, doRelu = flags & 2;
    int rowBase = m0 + 16 * w + ((lane >> 4) << 3);
#pragma unroll
    for (int s = 0; s < 4; ++s) {
        int col = n0 + s * 16 + (lane & 15);
        float b  = bias[col];
        float ga = doBN ? (BN_S * gam[col]) : 1.0f;
        float be = doBN ? bet[col] : 0.0f;
#pragma unroll
        for (int r = 0; r < 8; ++r) {
            float v = acc[s][r] + b;
            if (doBN) v = v * ga + be;
            if (doRelu) v = fmaxf(v, 0.0f);
            out[(size_t)(rowBase + r) * N + col] = v;
        }
    }
}

// one-hot @ le_w1 row gather: l1[i][h] = relu(bn(le_w1[y[i]][h] + b1[h]))
__global__ void k_label1(const int* __restrict__ y, const float* __restrict__ w1,
                         const float* __restrict__ b1, const float* __restrict__ g1,
                         const float* __restrict__ be1, float* __restrict__ out, int n) {
    int idx = blockIdx.x * blockDim.x + threadIdx.x;
    if (idx >= n * HD) return;
    int i = idx >> 8, h = idx & 255;
    float v = w1[y[i] * HD + h] + b1[h];
    v = v * (BN_S * g1[h]) + be1[h];
    out[idx] = fmaxf(v, 0.0f);
}

// ---------------------------------------------------------------------------
// Flash attention: softmax(Q K^T * 0.1) V, H=256. 64 Q-rows per WG, 64-key
// blocks, online softmax. Q/K/Vt tiles filled by async global->LDS b128 copies
// of pre-converted bf16 tensors (Vt global layout [H][NN]).
// ---------------------------------------------------------------------------
#define LDQ 264
#define LDK 264
#define LDV 72
#define LDP 72
#define LDSS 65
#define ATTN_SMEM_BYTES 132096

__global__ __launch_bounds__(256) void k_attn(const unsigned short* __restrict__ Qbf,
                                              const unsigned short* __restrict__ Kbf,
                                              const unsigned short* __restrict__ Vtb,
                                              float* __restrict__ Og, int nRows) {
    extern __shared__ unsigned short smem_us[];
    unsigned short* Qs = smem_us;             // 64*264
    unsigned short* Ks = Qs + 64 * LDQ;       // 64*264
    unsigned short* Vt = Ks + 64 * LDK;       // 256*72 ([h][key])
    unsigned short* Ps = Vt + 256 * LDV;      // 64*72
    float* Sf   = (float*)(smem_us + 56832);  // 64*65
    float* red  = Sf + 64 * LDSS;             // 256
    float* m_i  = red + 256;                  // 64
    float* l_i  = m_i + 64;                   // 64
    float* sc_s = l_i + 64;                   // 64

    int t = threadIdx.x, lane = t & 31, w = t >> 5;
    int q0 = blockIdx.x * 64;
    int mq = w & 3, nh = w >> 2;        // O mapping
    int ms = w & 3, nsB = (w >> 2) * 2; // S mapping
    int row = t >> 2, seg = t & 3;      // softmax mapping

    unsigned QsB = (unsigned)(unsigned long long)Qs;
    unsigned KsB = (unsigned)(unsigned long long)Ks;
    unsigned VtB = (unsigned)(unsigned long long)Vt;

    // ---- async-load Q tile once: 64 rows x 512B, one b128/lane per row ----
    {
        const unsigned short* gq = Qbf + (size_t)q0 * HD;
#pragma unroll
        for (int j = 0; j < 8; ++j) {
            int r = w * 8 + j;
            async_load_b128(QsB + (unsigned)(r * LDQ * 2) + (unsigned)(lane * 16),
                            gq + (size_t)r * HD + lane * 8);
        }
    }
    if (t < 64) { m_i[t] = -INFINITY; l_i[t] = 0.0f; }

    f32x8 o[8];
#pragma unroll
    for (int i = 0; i < 8; ++i) o[i] = (f32x8){0.f,0.f,0.f,0.f,0.f,0.f,0.f,0.f};
    wait_async0();
    __syncthreads();

    for (int kb = 0; kb < nRows; kb += 64) {
        // ---- async-load K rows + transposed V rows ----
        {
            const unsigned short* gk = Kbf + (size_t)kb * HD;
#pragma unroll
            for (int j = 0; j < 8; ++j) {
                int r = w * 8 + j;
                async_load_b128(KsB + (unsigned)(r * LDK * 2) + (unsigned)(lane * 16),
                                gk + (size_t)r * HD + lane * 8);
            }
#pragma unroll
            for (int j = 0; j < 8; ++j) { // Vt: 256 rows x 128B (4 rows/issue)
                int hr = w * 32 + j * 4 + (lane >> 3);
                int c8 = (lane & 7) * 8;
                async_load_b128(VtB + (unsigned)(hr * (LDV * 2)) + (unsigned)(c8 * 2),
                                Vtb + (size_t)hr * NN + kb + c8);
            }
        }
        wait_async0();
        __syncthreads();

        // ---- S = Q @ K^T (two 16x16 subtiles per wave) ----
        f32x8 s0 = {0.f,0.f,0.f,0.f,0.f,0.f,0.f,0.f};
        f32x8 s1 = s0;
#pragma unroll
        for (int k0 = 0; k0 < HD; k0 += 32) {
            bf16x16 fa  = frag_a(Qs, ms * 16, LDQ, k0, lane);
            bf16x16 fb0 = frag_b(Ks, nsB * 16, LDK, k0, lane);
            bf16x16 fb1 = frag_b(Ks, (nsB + 1) * 16, LDK, k0, lane);
            s0 = wmma_bf16(fa, fb0, s0);
            s1 = wmma_bf16(fa, fb1, s1);
        }
        {
            int colB = nsB * 16 + (lane & 15);
            int rowB = ms * 16 + ((lane >> 4) << 3);
#pragma unroll
            for (int r = 0; r < 8; ++r) {
                Sf[(rowB + r) * LDSS + colB]      = s0[r] * 0.1f;
                Sf[(rowB + r) * LDSS + colB + 16] = s1[r] * 0.1f;
            }
        }
        __syncthreads();

        // ---- online softmax (4 threads per row) ----
        float lmax = -INFINITY;
#pragma unroll
        for (int c = 0; c < 16; ++c)
            lmax = fmaxf(lmax, Sf[row * LDSS + seg * 16 + c]);
        red[row * 4 + seg] = lmax;
        __syncthreads();
        if (seg == 0) {
            float m4 = fmaxf(fmaxf(red[row * 4], red[row * 4 + 1]),
                             fmaxf(red[row * 4 + 2], red[row * 4 + 3]));
            float mo = m_i[row];
            float mn = fmaxf(mo, m4);
            sc_s[row] = (mo == -INFINITY) ? 0.0f : __expf(mo - mn);
            m_i[row] = mn;
        }
        __syncthreads();
        {
            float mn = m_i[row];
            float lsum = 0.0f;
#pragma unroll
            for (int c = 0; c < 16; c += 2) {
                float p0 = __expf(Sf[row * LDSS + seg * 16 + c] - mn);
                float p1 = __expf(Sf[row * LDSS + seg * 16 + c + 1] - mn);
                *(unsigned int*)&Ps[row * LDP + seg * 16 + c] = pk_bf16(p0, p1);
                lsum += p0 + p1;
            }
            red[row * 4 + seg] = lsum;
        }
        __syncthreads();
        if (seg == 0)
            l_i[row] = l_i[row] * sc_s[row] + red[row * 4] + red[row * 4 + 1] +
                       red[row * 4 + 2] + red[row * 4 + 3];
        __syncthreads();

        // ---- rescale O, then O += P @ V ----
        {
            int rowB = mq * 16 + ((lane >> 4) << 3);
            float sc[8];
#pragma unroll
            for (int r = 0; r < 8; ++r) sc[r] = sc_s[rowB + r];
#pragma unroll
            for (int i = 0; i < 8; ++i)
#pragma unroll
                for (int r = 0; r < 8; ++r) o[i][r] *= sc[r];
        }
#pragma unroll
        for (int j0 = 0; j0 < 64; j0 += 32) {
            bf16x16 fa = frag_a(Ps, mq * 16, LDP, j0, lane);
#pragma unroll
            for (int i = 0; i < 8; ++i) {
                bf16x16 fb = frag_b(Vt, (nh * 8 + i) * 16, LDV, j0, lane);
                o[i] = wmma_bf16(fa, fb, o[i]);
            }
        }
        __syncthreads();
    }

    // ---- finalize: O /= l, write z ----
    {
        int rowB = mq * 16 + ((lane >> 4) << 3);
        float inv[8];
#pragma unroll
        for (int r = 0; r < 8; ++r) inv[r] = 1.0f / l_i[rowB + r];
#pragma unroll
        for (int i = 0; i < 8; ++i) {
            int col = (nh * 8 + i) * 16 + (lane & 15);
#pragma unroll
            for (int r = 0; r < 8; ++r)
                Og[(size_t)(q0 + rowB + r) * HD + col] = o[i][r] * inv[r];
        }
    }
}

// ---------------------------------------------------------------------------
// pooling + classifier
// ---------------------------------------------------------------------------
__global__ void k_pool_accum(const float* __restrict__ z, const int* __restrict__ batch,
                             float* __restrict__ sums, float* __restrict__ cnt, int n) {
    int idx = blockIdx.x * blockDim.x + threadIdx.x;
    if (idx >= n * HD) return;
    int i = idx >> 8, h = idx & 255;
    int g = batch[i];
    atomicAdd(&sums[g * HD + h], z[idx]);
    if (h == 0) atomicAdd(&cnt[g], 1.0f);
}

__global__ __launch_bounds__(256) void k_pool_norm(const float* __restrict__ sums,
                                                   const float* __restrict__ cnt,
                                                   float* __restrict__ rep,
                                                   float* __restrict__ rep_out) {
    __shared__ float redn[256];
    int g = blockIdx.x, t = threadIdx.x;
    float c = fmaxf(cnt[g], 1.0f);
    float v = sums[g * HD + t] / c;
    redn[t] = v * v;
    __syncthreads();
    for (int s = 128; s > 0; s >>= 1) {
        if (t < s) redn[t] += redn[t + s];
        __syncthreads();
    }
    float inv = rsqrtf(redn[0]);
    float rv = v * inv;
    rep[g * HD + t] = rv;
    rep_out[g * HD + t] = rv;
}

__global__ __launch_bounds__(256) void k_cls(const float* __restrict__ rep,
                                             const float* __restrict__ w1,
                                             const float* __restrict__ b1,
                                             const float* __restrict__ g,
                                             const float* __restrict__ be,
                                             const float* __restrict__ w2,
                                             const float* __restrict__ b2,
                                             float* __restrict__ pred,
                                             float* __restrict__ c1) {
    __shared__ float lg[NG * NC];
    int t = threadIdx.x;
    for (int o = t; o < NG * HD; o += 256) {
        int r = o >> 8, h = o & 255;
        float acc = 0.0f;
        for (int k = 0; k < HD; ++k) acc += rep[r * HD + k] * w1[k * HD + h];
        float v = (acc + b1[h]) * (BN_S * g[h]) + be[h];
        c1[o] = fmaxf(v, 0.0f);
    }
    __syncthreads();
    for (int o = t; o < NG * NC; o += 256) {
        int r = o / NC, cc = o - r * NC;
        float acc = 0.0f;
        for (int k = 0; k < HD; ++k) acc += c1[r * HD + k] * w2[k * NC + cc];
        lg[o] = acc + b2[cc];
    }
    __syncthreads();
    if (t < NG) {
        float mx = -INFINITY;
        for (int cc = 0; cc < NC; ++cc) mx = fmaxf(mx, lg[t * NC + cc]);
        float sum = 0.0f;
        for (int cc = 0; cc < NC; ++cc) sum += __expf(lg[t * NC + cc] - mx);
        float ls = logf(sum);
        for (int cc = 0; cc < NC; ++cc) pred[t * NC + cc] = lg[t * NC + cc] - mx - ls;
    }
}

// ---------------------------------------------------------------------------
extern "C" void kernel_launch(void* const* d_in, const int* in_sizes, int n_in,
                              void* d_out, int out_size, void* d_ws, size_t ws_size,
                              hipStream_t stream) {
    (void)in_sizes; (void)n_in; (void)out_size; (void)ws_size;
    const float* x      = (const float*)d_in[0];
    const int*   y      = (const int*)d_in[1];
    const int*   ei     = (const int*)d_in[2];
    const int*   src    = ei;
    const int*   dstp   = ei + EE;
    const int*   batch  = (const int*)d_in[3];
    const float* gw1 = (const float*)d_in[4],  *gb1 = (const float*)d_in[5];
    const float* gg1 = (const float*)d_in[6],  *gbe1= (const float*)d_in[7];
    const float* gw2 = (const float*)d_in[8],  *gb2 = (const float*)d_in[9];
    const float* lw1 = (const float*)d_in[10], *lb1 = (const float*)d_in[11];
    const float* lg1 = (const float*)d_in[12], *lbe1= (const float*)d_in[13];
    const float* lw2 = (const float*)d_in[14], *lb2 = (const float*)d_in[15];
    const float* lg2 = (const float*)d_in[16], *lbe2= (const float*)d_in[17];
    const float* lw3 = (const float*)d_in[18], *lb3 = (const float*)d_in[19];
    const float* wq  = (const float*)d_in[20], *bq  = (const float*)d_in[21];
    const float* wk  = (const float*)d_in[22], *bk  = (const float*)d_in[23];
    const float* wv  = (const float*)d_in[24], *bv  = (const float*)d_in[25];
    const float* cw1 = (const float*)d_in[26], *cb1 = (const float*)d_in[27];
    const float* cg  = (const float*)d_in[28], *cbe = (const float*)d_in[29];
    const float* cw2 = (const float*)d_in[30], *cb2 = (const float*)d_in[31];

    float* p = (float*)d_ws;
    float* DINV = p; p += NN;
    float* A1   = p; p += (size_t)NN * FIN;
    float* Hb   = p; p += (size_t)NN * HD;  // h, later reused as l1
    float* A2   = p; p += (size_t)NN * HD;  // agg2, later reused as l2
    float* FEA  = p; p += (size_t)NN * HD;
    float* LE   = p; p += (size_t)NN * HD;
    float* Qb   = p; p += (size_t)NN * HD;
    float* Kb   = p; p += (size_t)NN * HD;
    float* Vb   = p; p += (size_t)NN * HD;
    float* Zb   = p; p += (size_t)NN * HD;
    unsigned short* Qbf = (unsigned short*)p; p += (size_t)NN * HD / 2;
    unsigned short* Kbf = (unsigned short*)p; p += (size_t)NN * HD / 2;
    unsigned short* Vtb = (unsigned short*)p; p += (size_t)NN * HD / 2;
    float* SUMS = p; p += NG * HD;
    float* CNT  = p; p += NG;
    float* REP  = p; p += NG * HD;
    float* C1   = p; p += NG * HD;

    float* out_pred = (float*)d_out;
    float* out_rep  = out_pred + NG * NC;

    // degrees + normalization
    k_fill<<<NN / 256, 256, 0, stream>>>(DINV, 1.0f, NN);
    k_deg_edges<<<EE / 256, 256, 0, stream>>>(dstp, DINV, EE);
    k_rsqrt<<<NN / 256, 256, 0, stream>>>(DINV, NN);

    // GCN layer 1
    k_selfinit<<<(NN * FIN) / 256, 256, 0, stream>>>(x, DINV, A1, NN, FIN);
    k_scatter<<<(EE * (FIN / 4)) / 256, 256, 0, stream>>>(src, dstp, DINV, x, A1, EE, FIN / 4);
    k_gemm<<<dim3(NN / 128, HD / 64), 256, 0, stream>>>(A1, gw1, gb1, gg1, gbe1, Hb,
                                                        NN, FIN, HD, 3);
    // GCN layer 2
    k_selfinit<<<(NN * HD) / 256, 256, 0, stream>>>(Hb, DINV, A2, NN, HD);
    k_scatter<<<(EE * (HD / 4)) / 256, 256, 0, stream>>>(src, dstp, DINV, Hb, A2, EE, HD / 4);
    k_gemm<<<dim3(NN / 128, HD / 64), 256, 0, stream>>>(A2, gw2, gb2, nullptr, nullptr, FEA,
                                                        NN, HD, HD, 0);
    // label encoder (l1 -> Hb reuse, l2 -> A2 reuse)
    k_label1<<<(NN * HD) / 256, 256, 0, stream>>>(y, lw1, lb1, lg1, lbe1, Hb, NN);
    k_gemm<<<dim3(NN / 128, HD / 64), 256, 0, stream>>>(Hb, lw2, lb2, lg2, lbe2, A2,
                                                        NN, HD, HD, 3);
    k_gemm<<<dim3(NN / 128, HD / 64), 256, 0, stream>>>(A2, lw3, lb3, nullptr, nullptr, LE,
                                                        NN, HD, HD, 0);
    // Q/K/V projections
    k_gemm<<<dim3(NN / 128, HD / 64), 256, 0, stream>>>(LE,  wq, bq, nullptr, nullptr, Qb,
                                                        NN, HD, HD, 0);
    k_gemm<<<dim3(NN / 128, HD / 64), 256, 0, stream>>>(FEA, wk, bk, nullptr, nullptr, Kb,
                                                        NN, HD, HD, 0);
    k_gemm<<<dim3(NN / 128, HD / 64), 256, 0, stream>>>(FEA, wv, bv, nullptr, nullptr, Vb,
                                                        NN, HD, HD, 0);

    // pre-convert to bf16 (Q,K row-major; V transposed [H][NN]) for async path
    k_cvt_pk<<<(NN * HD / 2) / 256, 256, 0, stream>>>(Qb, Qbf, NN * HD / 2);
    k_cvt_pk<<<(NN * HD / 2) / 256, 256, 0, stream>>>(Kb, Kbf, NN * HD / 2);
    k_cvt_t<<<(NN * HD) / 256, 256, 0, stream>>>(Vb, Vtb);

    // flash attention
    hipFuncSetAttribute((const void*)k_attn, hipFuncAttributeMaxDynamicSharedMemorySize,
                        ATTN_SMEM_BYTES);
    k_attn<<<NN / 64, 256, ATTN_SMEM_BYTES, stream>>>(Qbf, Kbf, Vtb, Zb, NN);

    // per-graph mean pooling + L2 normalize
    k_fill<<<(NG * HD) / 256, 256, 0, stream>>>(SUMS, 0.0f, NG * HD);
    k_fill<<<1, 64, 0, stream>>>(CNT, 0.0f, NG);
    k_pool_accum<<<(NN * HD) / 256, 256, 0, stream>>>(Zb, batch, SUMS, CNT, NN);
    k_pool_norm<<<NG, 256, 0, stream>>>(SUMS, CNT, REP, out_rep);

    // classifier + log_softmax
    k_cls<<<1, 256, 0, stream>>>(REP, cw1, cb1, cg, cbe, cw2, cb2, out_pred, C1);
}
